// MultiheadAttentionLayer_87462714015850
// MI455X (gfx1250) — matrix operand order repaired and checked by reading, not compiled, not run on hardware
//
#include <hip/hip_runtime.h>

typedef _Float16 f16;
typedef __attribute__((ext_vector_type(4)))  float    v4f;
typedef __attribute__((ext_vector_type(4)))  _Float16 v4h;
typedef __attribute__((ext_vector_type(8)))  _Float16 v8h;
typedef __attribute__((ext_vector_type(16))) _Float16 v16h;
typedef __attribute__((ext_vector_type(8)))  float    v8f;
typedef __attribute__((ext_vector_type(4)))  int      v4i;

#define D_MODEL 1024
#define NHEADS  8
#define DHEAD   128
#define SEQ     1024
#define BATCH   16
#define MROWS   (BATCH * SEQ)   // 16384

#if defined(__has_builtin)
#if __has_builtin(__builtin_amdgcn_global_load_async_to_lds_b128)
#define HAVE_ASYNC_LDS 1
#endif
#endif
#ifndef HAVE_ASYNC_LDS
#define HAVE_ASYNC_LDS 0
#endif

static __device__ __forceinline__ void copy16_to_lds(const f16* g, f16* l) {
#if HAVE_ASYNC_LDS
  __builtin_amdgcn_global_load_async_to_lds_b128((v4i*)g, (v4i*)l, 0, 0);
#else
  *(v8h*)l = *(const v8h*)g;
#endif
}

static __device__ __forceinline__ void wait_async0() {
#if defined(__has_builtin)
#if __has_builtin(__builtin_amdgcn_s_wait_asynccnt)
  __builtin_amdgcn_s_wait_asynccnt(0);
#else
  asm volatile("s_wait_asynccnt 0x0" ::: "memory");
#endif
#else
  asm volatile("s_wait_asynccnt 0x0" ::: "memory");
#endif
}

static __device__ __forceinline__ v16h cat16(v8h lo, v8h hi) {
  return __builtin_shufflevector(lo, hi, 0,1,2,3,4,5,6,7,8,9,10,11,12,13,14,15);
}

// ---------------------------------------------------------------- cast f32->f16 (x4 vectorized)
__global__ void cast_f32_f16_v4(const v4f* __restrict__ in, v4h* __restrict__ out, int n4) {
  int i = blockIdx.x * blockDim.x + threadIdx.x;
  int stride = gridDim.x * blockDim.x;
  for (; i < n4; i += stride) {
    v4f x = in[i];
    v4h y;
#pragma unroll
    for (int e = 0; e < 4; ++e) y[e] = (f16)x[e];
    out[i] = y;
  }
}

// ---------------------------------------------------------------- QKV projection
// C[m][n] = relu( sum_k A[m][k] * W[n][k] + bias[n] ), C stored f16.
// A: [MROWS][1024] f16, W: [1024][1024] f16 (PyTorch layout: row n, col k).
__global__ __launch_bounds__(256) void proj_relu_gemm(
    const f16* __restrict__ A, const f16* __restrict__ W,
    const float* __restrict__ bias, f16* __restrict__ C) {
  __shared__ __align__(16) f16 lA[2][128][32];   // [m][k] tile, double-buffered
  __shared__ __align__(16) f16 lB[2][128][32];   // [n][k] tile (N-major == W layout)
  const int m0 = blockIdx.x * 128;
  const int n0 = blockIdx.y * 128;
  const int tid  = threadIdx.x;
  const int lane = tid & 31, wave = tid >> 5;
  const int lrow = lane & 15, hlf = lane >> 4;
  const int wm = wave >> 1, wn = wave & 1;       // 4x2 wave grid: 32x64 per wave

  v8f acc[2][4] = {};

  auto stage = [&](int ks, int buf) {
    const int k0 = ks * 32;
#pragma unroll
    for (int i = 0; i < 2; ++i) {
      int c = tid + i * 256;            // 512 chunks of 8 halves per matrix
      int row = c >> 2, seg = c & 3;
      copy16_to_lds(A + (size_t)(m0 + row) * D_MODEL + k0 + seg * 8, &lA[buf][row][seg * 8]);
      copy16_to_lds(W + (size_t)(n0 + row) * D_MODEL + k0 + seg * 8, &lB[buf][row][seg * 8]);
    }
  };

  stage(0, 0);
  for (int ks = 0; ks < 32; ++ks) {
    const int buf = ks & 1;
    wait_async0();          // my async copies for tile ks have landed
    __syncthreads();        // everyone's have landed; prev-iter reads of buf^1 done
    if (ks + 1 < 32) stage(ks + 1, buf ^ 1);   // overlap next tile's DMA with compute

    v16h af[2], bf[4];
#pragma unroll
    for (int mi = 0; mi < 2; ++mi) {
      const f16* base = &lA[buf][wm * 32 + mi * 16 + lrow][0];
      af[mi] = cat16(*(const v8h*)(base + 8 * hlf), *(const v8h*)(base + 16 + 8 * hlf));
    }
#pragma unroll
    for (int ni = 0; ni < 4; ++ni) {
      const f16* base = &lB[buf][wn * 64 + ni * 16 + lrow][0];
      bf[ni] = cat16(*(const v8h*)(base + 8 * hlf), *(const v8h*)(base + 16 + 8 * hlf));
    }
#pragma unroll
    for (int mi = 0; mi < 2; ++mi)
#pragma unroll
      for (int ni = 0; ni < 4; ++ni)
        acc[mi][ni] = __builtin_amdgcn_wmma_f32_16x16x32_f16(
            false, af[mi], false, bf[ni], (short)0, acc[mi][ni], false, false);
  }

#pragma unroll
  for (int mi = 0; mi < 2; ++mi)
#pragma unroll
    for (int ni = 0; ni < 4; ++ni) {
      int gn = n0 + wn * 64 + ni * 16 + lrow;
      float bs = bias[gn];
#pragma unroll
      for (int v = 0; v < 8; ++v) {
        int gm = m0 + wm * 32 + mi * 16 + v + 8 * hlf;
        float x = acc[mi][ni][v] + bs;
        C[(size_t)gm * D_MODEL + gn] = (f16)(x > 0.f ? x : 0.f);
      }
    }
}

// ---------------------------------------------------------------- flash attention
// Per block: one (b,h), 128 query rows. 8 waves x 16 rows. Online softmax f32.
__global__ __launch_bounds__(256) void flash_attn(
    const f16* __restrict__ Q, const f16* __restrict__ K, const f16* __restrict__ V,
    float* __restrict__ O) {
  __shared__ __align__(16) f16 lK[2][32][128];    // [t][d], async double-buffered
  __shared__ __align__(16) f16 lVt[2][128][32];   // [d][t] (transposed at stage time)
  __shared__ __align__(16) f16 lP[8][16][32];     // per-wave P relayout buffer
  const int bh = blockIdx.x;
  const int b = bh >> 3, h = bh & 7;
  const int q0 = blockIdx.y * 128;
  const int tid  = threadIdx.x;
  const int lane = tid & 31, wave = tid >> 5;
  const int lrow = lane & 15, hlf = lane >> 4;
  const size_t rowbase = (size_t)b * SEQ;
  const int col0 = h * DHEAD;

  const f16* Qp = Q + col0;
  const f16* Kp = K + col0;
  const f16* Vp = V + col0;

  // Q fragments for this wave's 16 rows (kept in registers)
  v16h qf[4];
  {
    const f16* p = Qp + (rowbase + q0 + wave * 16 + lrow) * (size_t)D_MODEL;
#pragma unroll
    for (int kk = 0; kk < 4; ++kk) {
      const f16* pk = p + kk * 32;
      qf[kk] = cat16(*(const v8h*)(pk + 8 * hlf), *(const v8h*)(pk + 16 + 8 * hlf));
    }
  }

  v8f acc[8] = {};
  float m_run[8], l_run[8];
#pragma unroll
  for (int v = 0; v < 8; ++v) { m_run[v] = -1e30f; l_run[v] = 0.f; }
  const float SCALE = 0.08838834764831845f;  // 1/sqrt(128)

  auto stageKV = [&](int kt, int buf) {
    const int t0 = kt * 32;
#pragma unroll
    for (int i = 0; i < 2; ++i) {
      int c = tid + i * 256;            // 512 chunks: 32 rows x 16 segs
      int row = c >> 4, seg = c & 15;
      const size_t gr = (rowbase + t0 + row) * (size_t)D_MODEL + seg * 8;
      copy16_to_lds(Kp + gr, &lK[buf][row][seg * 8]);     // async direct to LDS
      v8h dv = *(const v8h*)(Vp + gr);                    // V needs transpose: via VGPR
#pragma unroll
      for (int e = 0; e < 8; ++e) lVt[buf][seg * 8 + e][row] = dv[e];
    }
  };

  stageKV(0, 0);
  for (int kt = 0; kt < 32; ++kt) {
    const int buf = kt & 1;
    wait_async0();
    __syncthreads();
    if (kt + 1 < 32) stageKV(kt + 1, buf ^ 1);

    // scores for two 16-key subtiles
    v8f s0 = {}, s1 = {};
#pragma unroll
    for (int kk = 0; kk < 4; ++kk) {
      const f16* b0 = &lK[buf][lrow][kk * 32];
      v16h bf0 = cat16(*(const v8h*)(b0 + 8 * hlf), *(const v8h*)(b0 + 16 + 8 * hlf));
      s0 = __builtin_amdgcn_wmma_f32_16x16x32_f16(false, qf[kk], false, bf0, (short)0, s0, false, false);
      const f16* b1 = &lK[buf][16 + lrow][kk * 32];
      v16h bf1 = cat16(*(const v8h*)(b1 + 8 * hlf), *(const v8h*)(b1 + 16 + 8 * hlf));
      s1 = __builtin_amdgcn_wmma_f32_16x16x32_f16(false, qf[kk], false, bf1, (short)0, s1, false, false);
    }

    // online softmax: rows live in VGPR index, 16 key-cols live across a 16-lane half
    float mx[8], al[8], rs[8];
#pragma unroll
    for (int v = 0; v < 8; ++v) {
      s0[v] *= SCALE; s1[v] *= SCALE;
      mx[v] = fmaxf(s0[v], s1[v]);
    }
#pragma unroll
    for (int o = 8; o >= 1; o >>= 1)
#pragma unroll
      for (int v = 0; v < 8; ++v)
        mx[v] = fmaxf(mx[v], __shfl_xor(mx[v], o, 32));
#pragma unroll
    for (int v = 0; v < 8; ++v) {
      float nm = fmaxf(m_run[v], mx[v]);
      al[v] = __expf(m_run[v] - nm);
      m_run[v] = nm;
      s0[v] = __expf(s0[v] - nm);
      s1[v] = __expf(s1[v] - nm);
      rs[v] = s0[v] + s1[v];
    }
#pragma unroll
    for (int o = 8; o >= 1; o >>= 1)
#pragma unroll
      for (int v = 0; v < 8; ++v)
        rs[v] += __shfl_xor(rs[v], o, 32);
#pragma unroll
    for (int v = 0; v < 8; ++v) l_run[v] = l_run[v] * al[v] + rs[v];
#pragma unroll
    for (int f = 0; f < 8; ++f)
#pragma unroll
      for (int v = 0; v < 8; ++v) acc[f][v] *= al[v];

    // relayout P (C-layout -> A-fragment layout) through per-wave LDS
#pragma unroll
    for (int v = 0; v < 8; ++v) {
      int r = v + 8 * hlf;
      lP[wave][r][lrow]      = (f16)s0[v];
      lP[wave][r][16 + lrow] = (f16)s1[v];
    }
    asm volatile("s_wait_dscnt 0x0" ::: "memory");
    const f16* pb = &lP[wave][lrow][0];
    v16h pf = cat16(*(const v8h*)(pb + 8 * hlf), *(const v8h*)(pb + 16 + 8 * hlf));

    // O += P x V  (B fragments from transposed V tile are contiguous)
#pragma unroll
    for (int f = 0; f < 8; ++f) {
      const f16* vb = &lVt[buf][f * 16 + lrow][0];
      v16h vf = cat16(*(const v8h*)(vb + 8 * hlf), *(const v8h*)(vb + 16 + 8 * hlf));
      acc[f] = __builtin_amdgcn_wmma_f32_16x16x32_f16(false, pf, false, vf, (short)0, acc[f], false, false);
    }
  }

#pragma unroll
  for (int f = 0; f < 8; ++f) {
    int gn = col0 + f * 16 + lrow;
#pragma unroll
    for (int v = 0; v < 8; ++v) {
      size_t gm = rowbase + q0 + wave * 16 + v + 8 * hlf;
      O[gm * D_MODEL + gn] = acc[f][v] / l_run[v];
    }
  }
}

// ---------------------------------------------------------------- residual + mask + LayerNorm
static __device__ __forceinline__ float block_sum(float v, float* red) {
  const int lane = threadIdx.x & 31, w = threadIdx.x >> 5;
#pragma unroll
  for (int o = 16; o >= 1; o >>= 1) v += __shfl_xor(v, o, 32);
  if (lane == 0) red[w] = v;
  __syncthreads();
  float t = (threadIdx.x < 8) ? red[threadIdx.x] : 0.f;
  if (w == 0) {
#pragma unroll
    for (int o = 4; o >= 1; o >>= 1) t += __shfl_xor(t, o, 32);
    if (lane == 0) red[0] = t;
  }
  __syncthreads();
  float r = red[0];
  __syncthreads();
  return r;
}

__global__ __launch_bounds__(256) void resid_ln(
    const float* __restrict__ attnout, const float* __restrict__ queries,
    const float* __restrict__ gamma, const float* __restrict__ beta,
    float* __restrict__ out) {
  __shared__ float red[8];
  const int r = blockIdx.x;
  const int tid = threadIdx.x;
  const float* qrow = queries + (size_t)r * D_MODEL;
  const float* arow = attnout + (size_t)r * D_MODEL;

  float q[4], a[4], y[4];
  float asum = 0.f;
#pragma unroll
  for (int i = 0; i < 4; ++i) {
    int c = tid + i * 256;
    q[i] = qrow[c];
    a[i] = arow[c];
    asum += fabsf(q[i]);
  }
  float mask = (block_sum(asum, red) > 0.f) ? 1.f : 0.f;

  float s = 0.f;
#pragma unroll
  for (int i = 0; i < 4; ++i) { y[i] = a[i] * mask + q[i]; s += y[i]; }
  float mu = block_sum(s, red) * (1.f / D_MODEL);

  float vs = 0.f;
#pragma unroll
  for (int i = 0; i < 4; ++i) { float d = y[i] - mu; vs += d * d; }
  float var = block_sum(vs, red) * (1.f / D_MODEL);
  float inv = rsqrtf(var + 1e-8f);

#pragma unroll
  for (int i = 0; i < 4; ++i) {
    int c = tid + i * 256;
    out[(size_t)r * D_MODEL + c] = (y[i] - mu) * inv * gamma[c] + beta[c];
  }
}

// ---------------------------------------------------------------- launch
extern "C" void kernel_launch(void* const* d_in, const int* in_sizes, int n_in,
                              void* d_out, int out_size, void* d_ws, size_t ws_size,
                              hipStream_t stream) {
  (void)in_sizes; (void)n_in; (void)out_size; (void)ws_size;
  const float* queries = (const float*)d_in[0];
  const float* keys    = (const float*)d_in[1];
  const float* Wq      = (const float*)d_in[2];
  const float* bq      = (const float*)d_in[3];
  const float* Wk      = (const float*)d_in[4];
  const float* bk      = (const float*)d_in[5];
  const float* Wv      = (const float*)d_in[6];
  const float* bv      = (const float*)d_in[7];
  const float* gamma   = (const float*)d_in[8];
  const float* beta    = (const float*)d_in[9];
  float* out = (float*)d_out;

  char* ws = (char*)d_ws;
  size_t off = 0;
  auto alloc = [&](size_t bytes) -> void* {
    void* p = ws + off;
    off += (bytes + 255) & ~(size_t)255;
    return p;
  };
  const size_t XB = (size_t)MROWS * D_MODEL * sizeof(f16);
  const size_t WB = (size_t)D_MODEL * D_MODEL * sizeof(f16);
  f16* Xq  = (f16*)alloc(XB);
  f16* Xk  = (f16*)alloc(XB);
  f16* Wqh = (f16*)alloc(WB);
  f16* Wkh = (f16*)alloc(WB);
  f16* Wvh = (f16*)alloc(WB);
  f16* Qb  = (f16*)alloc(XB);
  f16* Kb  = (f16*)alloc(XB);
  f16* Vb  = (f16*)alloc(XB);
  float* attn = (float*)alloc((size_t)MROWS * D_MODEL * sizeof(float));

  const int NX4 = MROWS * D_MODEL / 4;
  const int NW4 = D_MODEL * D_MODEL / 4;
  cast_f32_f16_v4<<<4096, 256, 0, stream>>>((const v4f*)queries, (v4h*)Xq, NX4);
  cast_f32_f16_v4<<<4096, 256, 0, stream>>>((const v4f*)keys,    (v4h*)Xk, NX4);
  cast_f32_f16_v4<<<512,  256, 0, stream>>>((const v4f*)Wq, (v4h*)Wqh, NW4);
  cast_f32_f16_v4<<<512,  256, 0, stream>>>((const v4f*)Wk, (v4h*)Wkh, NW4);
  cast_f32_f16_v4<<<512,  256, 0, stream>>>((const v4f*)Wv, (v4h*)Wvh, NW4);

  dim3 gp(MROWS / 128, D_MODEL / 128);  // 128 x 8
  proj_relu_gemm<<<gp, 256, 0, stream>>>(Xq, Wqh, bq, Qb);
  proj_relu_gemm<<<gp, 256, 0, stream>>>(Xk, Wkh, bk, Kb);
  proj_relu_gemm<<<gp, 256, 0, stream>>>(Xk, Wvh, bv, Vb);

  dim3 ga(BATCH * NHEADS, SEQ / 128);   // 128 x 8
  flash_attn<<<ga, 256, 0, stream>>>(Qb, Kb, Vb, attn);

  resid_ln<<<MROWS, 256, 0, stream>>>(attn, queries, gamma, beta, out);
}
